// Mix3DEmbeddingV6_6949257085362
// MI455X (gfx1250) — matrix operand order repaired and checked by reading, not compiled, not run on hardware
//
#include <hip/hip_runtime.h>
#include <hip/hip_bf16.h>
#include <math.h>

typedef __attribute__((ext_vector_type(16))) _Float16 v16h;
typedef __attribute__((ext_vector_type(8)))  float    v8f;

#define NB 4
#define NL 2048
#define NN (NB * NL)        // 8192 nodes
#define TT 512
#define NTE (NN + 16)       // TE rows (row NN == t=0), padded to 16

union AFrag {
    v16h v;
    _Float16 h[16];
    unsigned u[8];
};

// ---------------------------------------------------------------------------
// WMMA fragment loaders (wave32, v_wmma_f32_16x16x32_f16 layouts per CDNA5 ISA)
// A: 16x32 f16.  lanes0-15: M=lane, K-base 0; lanes16-31: M=lane-16, K-base 8.
//    VGPR v holds K = base + 2v + (v>=4 ? 8 : 0) (+0,+1 in the two halves).
// B: 32x16 f16.  lanes0-15: N=lane, K-base 0; lanes16-31: N=lane-16, K-base 8,
//    same K order as A (VGPR v: K = base + 2v + (v>=4 ? 8 : 0)).
// C/D: 16x16 f32. VGPR r: M = r + 8*(lane>=16), N = lane&15.
// ---------------------------------------------------------------------------
__device__ inline v16h load_a_frag(const _Float16* A, int lda, int m0, int k0, int lane) {
    int m  = lane & 15;
    int kb = (lane >> 4) << 3;          // 0 or 8
    const _Float16* row = A + (size_t)(m0 + m) * lda + k0 + kb;
    AFrag f;
#pragma unroll
    for (int v = 0; v < 8; ++v) {
        int koff = 2 * v + ((v >= 4) ? 8 : 0);
        f.u[v] = *(const unsigned*)(row + koff);
    }
    return f.v;
}

// Bw is row-major (N x K) holding W so that B[k][n] = Bw[n*ldb + k]  (i.e. X @ W^T)
__device__ inline v16h load_b_frag(const _Float16* Bw, int ldb, int n0, int k0, int lane) {
    int n  = lane & 15;
    int kb = (lane >> 4) << 3;          // 0 or 8
    const _Float16* p = Bw + (size_t)(n0 + n) * ldb + k0 + kb;
    AFrag f;
#pragma unroll
    for (int v = 0; v < 8; ++v) f.u[v] = *(const unsigned*)(p + 2 * v + ((v >= 4) ? 8 : 0));
    return f.v;
}

__device__ inline v8f wmma_f16(v16h a, v16h b, v8f c) {
    return __builtin_amdgcn_wmma_f32_16x16x32_f16(false, a, false, b, (short)0, c, false, false);
}

__device__ inline v8f zero8() {
    v8f z = {0.f, 0.f, 0.f, 0.f, 0.f, 0.f, 0.f, 0.f};
    return z;
}

// ---------------------------------------------------------------------------
// Kernel 0: f32 -> f16 weight conversion
// ---------------------------------------------------------------------------
__global__ void k_cvt_f16(const float* in, _Float16* out, int n) {
    int i = blockIdx.x * blockDim.x + threadIdx.x;
    if (i < n) out[i] = (_Float16)in[i];
}

// ---------------------------------------------------------------------------
// Kernel 1: time-embedding table. Fused 2-layer MLP.
// TE[r] for r<NN: t=time_pos[r]; r>=NN: t=0  (row NN is the t0 row).
// Each block: 16 rows x full 512 cols; each wave owns 4 col tiles (16x64):
// A fragment loaded once per K-step, reused across 4 WMMAs (no spills).
// ---------------------------------------------------------------------------
__global__ void k_time_emb(const float* __restrict__ time_pos,
                           const _Float16* __restrict__ w1f, const float* __restrict__ b1,
                           const _Float16* __restrict__ w2f, const float* __restrict__ b2,
                           float* __restrict__ TE) {
    __shared__ _Float16 s_emb[16][TT];
    __shared__ _Float16 s_h[16][TT];
    __shared__ float s_t[16];

    int r0  = blockIdx.x * 16;
    int tid = threadIdx.x;
    if (tid < 16) {
        int gr = r0 + tid;
        s_t[tid] = (gr < NN) ? time_pos[gr] : 0.0f;
    }
    __syncthreads();

    const float cexp = -0.03597789207f;  // -ln(10000)/256
    for (int e = tid; e < 16 * TT; e += 256) {
        int rr = e >> 9, k = e & 511;
        float t    = s_t[rr];
        float freq = expf(cexp * (float)(k & 255));
        float ang  = t * freq;
        s_emb[rr][k] = (_Float16)((k < 256) ? sinf(ang) : cosf(ang));
    }
    __syncthreads();

    int wave  = tid >> 5, lane = tid & 31;
    int n     = lane & 15;
    int mhi   = (lane >> 4) << 3;
    int nbase = wave * 64;

    // layer 1: H = silu(emb @ w1^T + b1)
    {
        v8f acc[4] = {zero8(), zero8(), zero8(), zero8()};
#pragma unroll 1
        for (int ks = 0; ks < TT; ks += 32) {
            v16h a = load_a_frag(&s_emb[0][0], TT, 0, ks, lane);
#pragma unroll
            for (int c = 0; c < 4; ++c) {
                v16h b = load_b_frag(w1f, TT, nbase + c * 16, ks, lane);
                acc[c] = wmma_f16(a, b, acc[c]);
            }
        }
#pragma unroll
        for (int c = 0; c < 4; ++c) {
            int col = nbase + c * 16 + n;
            float bb = b1[col];
#pragma unroll
            for (int r = 0; r < 8; ++r) {
                float x = acc[c][r] + bb;
                float s = x / (1.0f + expf(-x));
                s_h[r + mhi][col] = (_Float16)s;
            }
        }
    }
    __syncthreads();

    // layer 2: TE = H @ w2^T + b2
    {
        v8f acc[4] = {zero8(), zero8(), zero8(), zero8()};
#pragma unroll 1
        for (int ks = 0; ks < TT; ks += 32) {
            v16h a = load_a_frag(&s_h[0][0], TT, 0, ks, lane);
#pragma unroll
            for (int c = 0; c < 4; ++c) {
                v16h b = load_b_frag(w2f, TT, nbase + c * 16, ks, lane);
                acc[c] = wmma_f16(a, b, acc[c]);
            }
        }
#pragma unroll
        for (int c = 0; c < 4; ++c) {
            int col = nbase + c * 16 + n;
            float bb = b2[col];
#pragma unroll
            for (int r = 0; r < 8; ++r) {
                TE[(size_t)(r0 + r + mhi) * TT + col] = acc[c][r] + bb;
            }
        }
    }
}

// ---------------------------------------------------------------------------
// Kernel 2: angle feature rows (49152 x 512 f16), fused:
//   base = am ? gelu(ang*w1) @ w2^T : unkang ;  += te_ang (gathered from TE)
// A fragments (K=128 -> 4 of them) built in registers from gelu values once,
// reused across all 32 col tiles.
// ---------------------------------------------------------------------------
__global__ void k_angle(const float* __restrict__ angle, const int* __restrict__ angle_mask,
                        const int* __restrict__ bond_mask, const int* __restrict__ mask_angle,
                        const int* __restrict__ aa, const float* __restrict__ ang_w1,
                        const _Float16* __restrict__ w2f, const float* __restrict__ unkang,
                        const float* __restrict__ TE, _Float16* __restrict__ A_ang) {
    __shared__ float s_w1[128];
    __shared__ float s_ang[16];
    __shared__ int   s_amv[16];
    __shared__ int   s_teRow[16];

    int R0  = blockIdx.x * 16;
    int tid = threadIdx.x;
    if (tid < 128) s_w1[tid] = ang_w1[tid];
    if (tid < 16) {
        int R    = R0 + tid;
        int b    = R / (6 * NL);
        int rem  = R % (6 * NL);
        int l    = rem / 6, c = rem % 6;
        int node = b * NL + l;
        int amv  = (c < 3) ? angle_mask[node * 3 + c] : bond_mask[node * 3 + (c - 3)];
        s_ang[tid] = amv ? angle[node * 6 + c] : 0.0f;
        s_amv[tid] = amv;
        // reference quirk: t_mask is tiled as n % L (not l!)
        int idx   = rem % NL;
        int node2 = b * NL + idx;
        int a2    = aa[node2];
        int tm    = mask_angle[node2] && (a2 != 0) && (a2 != 2);
        s_teRow[tid] = (tm && amv) ? node : NN;   // NN = t0 row
    }
    __syncthreads();

    int wave = tid >> 5, lane = tid & 31;
    int m_lane = lane & 15;
    int kb  = (lane >> 4) << 3;
    int mhi = (lane >> 4) << 3;
    float av = s_ang[m_lane];

    // Build the 4 K-step A fragments once (K=128)
    v16h afr[4];
#pragma unroll
    for (int ksi = 0; ksi < 4; ++ksi) {
        AFrag f;
#pragma unroll
        for (int i = 0; i < 16; ++i) {
            int v = i >> 1;
            int k = ksi * 32 + kb + 2 * v + ((v >= 4) ? 8 : 0) + (i & 1);
            float x = av * s_w1[k];
            float g = 0.5f * x * (1.0f + erff(x * 0.70710678118f));
            f.h[i] = (_Float16)g;
        }
        afr[ksi] = f.v;
    }

#pragma unroll 1
    for (int it = 0; it < 4; ++it) {
        int n0 = (wave + 8 * it) * 16;
        v8f acc = zero8();
#pragma unroll
        for (int ksi = 0; ksi < 4; ++ksi) {
            v16h b = load_b_frag(w2f, 128, n0, ksi * 32, lane);
            acc = wmma_f16(afr[ksi], b, acc);
        }
        int col = n0 + (lane & 15);
        float unk = unkang[col];
#pragma unroll
        for (int r = 0; r < 8; ++r) {
            int m = r + mhi;
            float base = s_amv[m] ? acc[r] : unk;
            float te   = TE[(size_t)s_teRow[m] * TT + col];
            A_ang[(size_t)(R0 + m) * TT + col] = (_Float16)(base + te);
        }
    }
}

// ---------------------------------------------------------------------------
// Kernel 3a: segment bounds (generic cumsum-based), serial per batch (tiny)
// ---------------------------------------------------------------------------
__global__ void k_seg(const int* __restrict__ aa, int* __restrict__ segStart,
                      int* __restrict__ segEnd) {
    if (threadIdx.x != 0) return;
    int b = blockIdx.x;
    const int* a = aa + b * NL;
    int* ss = segStart + b * NL;
    int* se = segEnd + b * NL;
    int start = 0;
    for (int l = 0; l < NL; ++l) {
        if (a[l] == 0) start = l;
        ss[l] = start;
    }
    int end = NL - 1;
    for (int l = NL - 1; l >= 0; --l) {
        se[l] = end;
        if (a[l] == 0) end = l - 1;
    }
}

// ---------------------------------------------------------------------------
// Kernel 3b: ca, |ca|^2, pos_p = ca @ pos_emb_w^T  (K=32)
// ---------------------------------------------------------------------------
__global__ void k_posp(const float* __restrict__ pos, const float* __restrict__ pos_emb_w,
                       float* __restrict__ ca, float* __restrict__ sq,
                       float* __restrict__ pos_p) {
    int tid = threadIdx.x;
    int m = blockIdx.x * 8 + (tid >> 5);
    int k = tid & 31;
    float c0 = pos[(size_t)m * 9 + 3 + 0];
    float c1 = pos[(size_t)m * 9 + 3 + 1];
    float c2 = pos[(size_t)m * 9 + 3 + 2];
    pos_p[(size_t)m * 32 + k] =
        c0 * pos_emb_w[k * 3 + 0] + c1 * pos_emb_w[k * 3 + 1] + c2 * pos_emb_w[k * 3 + 2];
    if (k == 0) {
        ca[m * 3 + 0] = c0; ca[m * 3 + 1] = c1; ca[m * 3 + 2] = c2;
        sq[m] = c0 * c0 + c1 * c1 + c2 * c2;
    }
}

// ---------------------------------------------------------------------------
// Kernel 3c: segment-restricted weighted sum: sum_dist[l,k] = sum_m w(l,m)*pos_p[m,k]
// One wave per row; lane = output k.
// ---------------------------------------------------------------------------
__global__ void k_sumdist(const float* __restrict__ ca, const float* __restrict__ sq,
                          const float* __restrict__ pos_p, const int* __restrict__ aa,
                          const int* __restrict__ segStart, const int* __restrict__ segEnd,
                          float* __restrict__ sum_dist) {
    int wave = threadIdx.x >> 5, lane = threadIdx.x & 31;
    int l = blockIdx.x * 8 + wave;
    float* outp = sum_dist + (size_t)l * 32;
    int al = aa[l];
    if (al == 0 || al == 2) { outp[lane] = 0.0f; return; }
    int b  = l >> 11;
    int s0 = segStart[l], s1 = segEnd[l];
    float c0 = ca[l * 3], c1 = ca[l * 3 + 1], c2 = ca[l * 3 + 2];
    float sql = sq[l];
    float acc = 0.0f;
    for (int mm = s0; mm <= s1; ++mm) {
        int m  = b * NL + mm;
        int am = aa[m];
        if (am != 0 && am != 2) {
            float dot = c0 * ca[m * 3] + c1 * ca[m * 3 + 1] + c2 * ca[m * 3 + 2];
            float d2  = fmaxf(sql + sq[m] - 2.0f * dot, 0.0f);
            float w   = 1.0f / (sqrtf(d2) + 1.0f);
            acc += w * pos_p[(size_t)m * 32 + lane];
        }
    }
    outp[lane] = acc;
}

// ---------------------------------------------------------------------------
// Kernel 3d: pos_feat rows (8192 x 512 f16), fused:
//   base = valid ? sum_dist @ pfe^T : unkpos ;  += te_pos (gathered)
// K=32 -> exactly one WMMA per tile; A fragment reused across 4 col tiles.
// ---------------------------------------------------------------------------
__global__ void k_posfeat(const float* __restrict__ sum_dist, const _Float16* __restrict__ pfef,
                          const int* __restrict__ aa, const int* __restrict__ mask_angle,
                          const float* __restrict__ unkpos, const float* __restrict__ TE,
                          _Float16* __restrict__ A_pos) {
    __shared__ int s_valid[16];
    __shared__ int s_teRow[16];
    int R0  = blockIdx.x * 16;
    int tid = threadIdx.x;
    if (tid < 16) {
        int R = R0 + tid;
        int a = aa[R];
        int v = (a != 0) && (a != 2);
        s_valid[tid] = v;
        s_teRow[tid] = (mask_angle[R] && v) ? R : NN;
    }
    __syncthreads();

    int wave = tid >> 5, lane = tid & 31;
    int m_lane = lane & 15;
    int kb  = (lane >> 4) << 3;
    int mhi = (lane >> 4) << 3;

    // A fragment from f32 sum_dist (K=32)
    AFrag f;
    const float* row = sum_dist + (size_t)(R0 + m_lane) * 32;
#pragma unroll
    for (int i = 0; i < 16; ++i) {
        int v = i >> 1;
        int koff = kb + 2 * v + ((v >= 4) ? 8 : 0) + (i & 1);
        f.h[i] = (_Float16)row[koff];
    }

#pragma unroll 1
    for (int it = 0; it < 4; ++it) {
        int n0 = (wave + 8 * it) * 16;
        v8f acc = zero8();
        v16h b = load_b_frag(pfef, 32, n0, 0, lane);
        acc = wmma_f16(f.v, b, acc);
        int col = n0 + (lane & 15);
        float unk = unkpos[col];
#pragma unroll
        for (int r = 0; r < 8; ++r) {
            int m = r + mhi;
            float base = s_valid[m] ? acc[r] : unk;
            float te   = TE[(size_t)s_teRow[m] * TT + col];
            A_pos[(size_t)(R0 + m) * TT + col] = (_Float16)(base + te);
        }
    }
}

// ---------------------------------------------------------------------------
// Kernel 4: projection GEMM  out[:, colOffset:colOffset+512] = A(8192xK) @ Bw^T
// grid = M/16 blocks; each wave owns 4 col tiles (16x64): A fragment loaded
// once per K-step and reused across 4 WMMAs.
// ---------------------------------------------------------------------------
__global__ void k_proj(const _Float16* __restrict__ A, const _Float16* __restrict__ Bw,
                       float* __restrict__ out, int K, int colOffset) {
    int wave = threadIdx.x >> 5, lane = threadIdx.x & 31;
    int R0    = blockIdx.x * 16;
    int nbase = wave * 64;
    v8f acc[4] = {zero8(), zero8(), zero8(), zero8()};
#pragma unroll 1
    for (int ks = 0; ks < K; ks += 32) {
        v16h a = load_a_frag(A, K, R0, ks, lane);
#pragma unroll
        for (int c = 0; c < 4; ++c) {
            v16h b = load_b_frag(Bw, K, nbase + c * 16, ks, lane);
            acc[c] = wmma_f16(a, b, acc[c]);
        }
    }
    int n   = lane & 15;
    int mhi = (lane >> 4) << 3;
#pragma unroll
    for (int c = 0; c < 4; ++c) {
        int col = nbase + c * 16 + n;
#pragma unroll
        for (int r = 0; r < 8; ++r) {
            out[(size_t)(R0 + r + mhi) * 1024 + colOffset + col] = acc[c][r];
        }
    }
}

// ---------------------------------------------------------------------------
// Kernel 5: cls / eos / padding overrides
// ---------------------------------------------------------------------------
__global__ void k_final(const int* __restrict__ aa, const int* __restrict__ pad,
                        const float* __restrict__ cls_w, const float* __restrict__ eos_w,
                        float* __restrict__ out) {
    int R = blockIdx.x;
    int a = aa[R];
    int p = pad[R];
    float* o = out + (size_t)R * 1024;
    for (int j = threadIdx.x; j < 1024; j += blockDim.x) {
        float v = o[j];
        if (a == 0) v = cls_w[j];
        if (a == 2) v = eos_w[j];
        if (p)      v = 0.0f;
        o[j] = v;
    }
}

// ---------------------------------------------------------------------------
extern "C" void kernel_launch(void* const* d_in, const int* in_sizes, int n_in,
                              void* d_out, int out_size, void* d_ws, size_t ws_size,
                              hipStream_t stream) {
    const float* pos        = (const float*)d_in[0];
    const float* angle      = (const float*)d_in[1];
    const int*   padding    = (const int*)d_in[2];
    const int*   mask_angle = (const int*)d_in[5];
    const int*   angle_mask = (const int*)d_in[6];
    const int*   bond_mask  = (const int*)d_in[7];
    const float* time_pos   = (const float*)d_in[8];
    const int*   aa         = (const int*)d_in[10];
    const float* pos_emb_w  = (const float*)d_in[11];
    const float* pfe_w      = (const float*)d_in[12];
    const float* ang_w1     = (const float*)d_in[13];
    const float* ang_w2     = (const float*)d_in[14];
    const float* t_w1       = (const float*)d_in[15];
    const float* t_b1       = (const float*)d_in[16];
    const float* t_w2       = (const float*)d_in[17];
    const float* t_b2       = (const float*)d_in[18];
    const float* ang_proj_w = (const float*)d_in[19];
    const float* pos_proj_w = (const float*)d_in[20];
    const float* cls_w      = (const float*)d_in[21];
    const float* eos_w      = (const float*)d_in[22];
    const float* unkang_w   = (const float*)d_in[23];
    const float* unkpos_w   = (const float*)d_in[24];
    float* out = (float*)d_out;

    char* wsp = (char*)d_ws;
    auto take = [&](size_t bytes) -> void* {
        void* p = (void*)wsp;
        wsp += (bytes + 255) & ~(size_t)255;
        return p;
    };
    float*    TE        = (float*)take((size_t)NTE * TT * 4);          // 16.8 MB
    _Float16* A_ang     = (_Float16*)take((size_t)NN * 6 * TT * 2);    // 50.3 MB
    _Float16* A_pos     = (_Float16*)take((size_t)NN * TT * 2);        //  8.4 MB
    float*    sum_dist  = (float*)take((size_t)NN * 32 * 4);
    float*    pos_p     = (float*)take((size_t)NN * 32 * 4);
    float*    ca        = (float*)take((size_t)NN * 3 * 4);
    float*    sq        = (float*)take((size_t)NN * 4);
    int*      segStart  = (int*)take((size_t)NN * 4);
    int*      segEnd    = (int*)take((size_t)NN * 4);
    _Float16* w1f       = (_Float16*)take((size_t)TT * TT * 2);
    _Float16* w2f       = (_Float16*)take((size_t)TT * TT * 2);
    _Float16* angw2f    = (_Float16*)take((size_t)TT * 128 * 2);
    _Float16* pfef      = (_Float16*)take((size_t)TT * 32 * 2);
    _Float16* angprojf  = (_Float16*)take((size_t)TT * 3072 * 2);
    _Float16* posprojf  = (_Float16*)take((size_t)TT * TT * 2);

    // 0) weight conversions
    auto cvt = [&](const float* src, _Float16* dst, int n) {
        k_cvt_f16<<<(n + 255) / 256, 256, 0, stream>>>(src, dst, n);
    };
    cvt(t_w1, w1f, TT * TT);
    cvt(t_w2, w2f, TT * TT);
    cvt(ang_w2, angw2f, TT * 128);
    cvt(pfe_w, pfef, TT * 32);
    cvt(ang_proj_w, angprojf, TT * 3072);
    cvt(pos_proj_w, posprojf, TT * TT);

    // 1) time-embedding table (8192 real rows + t0 row)
    k_time_emb<<<NTE / 16, 256, 0, stream>>>(time_pos, w1f, t_b1, w2f, t_b2, TE);

    // 2) angle feature rows -> A_ang (49152 x 512 f16 == 8192 x 3072)
    k_angle<<<(NN * 6) / 16, 256, 0, stream>>>(angle, angle_mask, bond_mask, mask_angle,
                                               aa, ang_w1, angw2f, unkang_w, TE, A_ang);

    // 3) position branch
    k_seg<<<NB, 32, 0, stream>>>(aa, segStart, segEnd);
    k_posp<<<NN / 8, 256, 0, stream>>>(pos, pos_emb_w, ca, sq, pos_p);
    k_sumdist<<<NN / 8, 256, 0, stream>>>(ca, sq, pos_p, aa, segStart, segEnd, sum_dist);
    k_posfeat<<<NN / 16, 256, 0, stream>>>(sum_dist, pfef, aa, mask_angle, unkpos_w, TE, A_pos);

    // 4) projections into the concatenated output
    k_proj<<<NN / 16, 256, 0, stream>>>(A_ang, angprojf, out, 3072, 0);
    k_proj<<<NN / 16, 256, 0, stream>>>(A_pos, posprojf, out, TT, 512);

    // 5) cls / eos / padding overrides
    k_final<<<NN, 256, 0, stream>>>(aa, padding, cls_w, eos_w, out);
}
